// EdgeQNetwork_89653147337560
// MI455X (gfx1250) — matrix operand order repaired and checked by reading, not compiled, not run on hardware
//
#include <hip/hip_runtime.h>

// ---------------------------------------------------------------------------
// EdgeQNetwork on MI455X (gfx1250), wave32 + WMMA.
//
// Algebra: the per-edge MLP has no activations between its three Linears, so
// it collapses to Wc = W3*W2*W1, bc = W3*(W2*b1+b2)+b3.  With h=[xi | xj-xi]:
//     h[e] = (Wa-Wb)@x[dst] + Wb@x[src] + bc = A[dst] + B[src] + bc
// and segment_max over dst:  agg[i] = A[i] + bc + max_{dst=i} B[src].
// This turns 180 GFLOP of [65536,510] edge GEMMs into [256,255] node GEMMs.
//
// Roofline: remaining cost is streaming l4_W (534MB) + out_W (267MB) fp32 at
// 23.3 TB/s ~= 34us -> bandwidth-bound float4 mat-vecs with non-temporal
// loads (don't evict the L2-resident B matrix / g vector / y).  Small GEMMs
// run on v_wmma_f32_16x16x32_f16; all WMMA operands are NT-shaped ([M,K] x
// [N,K]) so each fragment is exactly two global_load_b128 per lane (ISA
// 7.12.2 layouts are lane-contiguous in that orientation) -- no LDS staging.
// Workspace required: ~2.8 MB.
// ---------------------------------------------------------------------------

typedef __attribute__((ext_vector_type(16))) _Float16 v16h;
typedef __attribute__((ext_vector_type(8)))  _Float16 v8h;
typedef __attribute__((ext_vector_type(8)))  float    v8f;
typedef __attribute__((ext_vector_type(4)))  float    f4;

#define FDIM  255
#define NNODE 256
#define NEDGE 65536
#define FP    256    // padded F
#define F2P   512    // padded 2F
#define HDIM  2048
#define OUTN  32640
#define GVEC  65280  // NNODE * FDIM

// ---------------- edge counting sort (dst buckets) -------------------------

__global__ void k_zero256(int* counts, int* cursors) {
    int i = threadIdx.x;
    counts[i] = 0;
    cursors[i] = 0;
}

__global__ void k_hist(const int* __restrict__ dst, int* __restrict__ counts) {
    int e = blockIdx.x * blockDim.x + threadIdx.x;
    if (e < NEDGE) atomicAdd(&counts[dst[e]], 1);
}

__global__ void k_scan(const int* __restrict__ counts, int* __restrict__ offsets) {
    __shared__ int t[256];
    int i = threadIdx.x;
    int c = counts[i];
    t[i] = c;
    __syncthreads();
    for (int off = 1; off < 256; off <<= 1) {
        int v = (i >= off) ? t[i - off] : 0;
        __syncthreads();
        t[i] += v;
        __syncthreads();
    }
    offsets[i] = t[i] - c;          // exclusive prefix
    if (i == 255) offsets[256] = t[255];
}

__global__ void k_scatter(const int* __restrict__ src, const int* __restrict__ dst,
                          const int* __restrict__ offsets, int* __restrict__ cursors,
                          int* __restrict__ sorted_src) {
    int e = blockIdx.x * blockDim.x + threadIdx.x;
    if (e < NEDGE) {
        int d = dst[e];
        int p = atomicAdd(&cursors[d], 1);       // order within segment is
        sorted_src[offsets[d] + p] = src[e];     // irrelevant: max is exact
    }
}

// ---------------- fp32 -> padded fp16 (plain and transposed) ---------------

__global__ void k_cvt_pad(const float* __restrict__ src, int srows, int scols,
                          _Float16* __restrict__ dst, int ldd) {
    int r = blockIdx.x;
    for (int c = threadIdx.x; c < ldd; c += blockDim.x) {
        float v = (r < srows && c < scols) ? src[(size_t)r * scols + c] : 0.f;
        dst[(size_t)r * ldd + c] = (_Float16)v;
    }
}

// dst[r][c] = src[c][r]  (dst is [scols pad][srows pad])
__global__ void k_cvt_pad_t(const float* __restrict__ src, int srows, int scols,
                            _Float16* __restrict__ dst, int ldd) {
    int r = blockIdx.x;
    for (int c = threadIdx.x; c < ldd; c += blockDim.x) {
        float v = (r < scols && c < srows) ? src[(size_t)c * scols + r] : 0.f;
        dst[(size_t)r * ldd + c] = (_Float16)v;
    }
}

// ---------------- WMMA NT GEMM: C = A[M,K] * Bt[N,K]^T ---------------------
// One wave per 16x16 C tile.  Per CDNA5 ISA 7.12.2 (f16 16x16x32):
//   A frag lane: M = lane&15 ; elements 0..7  -> K = hi*8 + [0..8)
//                              elements 8..15 -> K = hi*8 + 16 + [0..8)
//   B frag lane: N = lane&15 ; same K order as A frag
//   (hi = lane>=16)  => each fragment = two 16B global_load_b128.
//   C frag: vgpr r -> M = hi*8 + r ; N = lane&15.
// All operands are zero-padded in workspace: no bounds checks, EXEC all-1s.

union V16 { v16h v; v8h h[2]; };

template <bool HALF_OUT, bool STORE_T>
__global__ void k_gemm_nt(const _Float16* __restrict__ A, const _Float16* __restrict__ Bt,
                          void* __restrict__ Cp, int K, int lda, int ldb, int ldc) {
    const int lane = threadIdx.x & 31;
    const int hi   = lane >> 4;
    const int row  = lane & 15;
    const int tn = blockIdx.x, tm = blockIdx.y;

    const _Float16* Ar = A  + (size_t)(tm * 16 + row) * lda + hi * 8;
    const _Float16* Br = Bt + (size_t)(tn * 16 + row) * ldb + hi * 8;

    v8f acc = {};
    for (int k0 = 0; k0 < K; k0 += 32) {
        V16 a, b;
        a.h[0] = *(const v8h*)(Ar + k0);
        a.h[1] = *(const v8h*)(Ar + k0 + 16);
        b.h[0] = *(const v8h*)(Br + k0);
        b.h[1] = *(const v8h*)(Br + k0 + 16);
        acc = __builtin_amdgcn_wmma_f32_16x16x32_f16(false, a.v, false, b.v,
                                                     (short)0, acc, false, false);
    }
#pragma unroll
    for (int r = 0; r < 8; ++r) {
        int m = tm * 16 + (hi << 3) + r;
        int n = tn * 16 + row;
        size_t idx = STORE_T ? (size_t)n * ldc + m : (size_t)m * ldc + n;
        if (HALF_OUT) ((_Float16*)Cp)[idx] = (_Float16)acc[r];
        else          ((float*)Cp)[idx]    = acc[r];
    }
}

// ---------------- bias composition: bc = W3*(W2*b1+b2)+b3 (fp32) -----------

__global__ void k_bias(const float* __restrict__ W2, const float* __restrict__ b1,
                       const float* __restrict__ b2, const float* __restrict__ W3,
                       const float* __restrict__ b3, float* __restrict__ bc) {
    __shared__ float tmp[256];
    int f = threadIdx.x;
    float t = 0.f;
    if (f < FDIM) {
        const float* w = W2 + (size_t)f * (2 * FDIM);
        for (int k = 0; k < 2 * FDIM; ++k) t += w[k] * b1[k];
        t += b2[f];
    }
    tmp[f] = t;
    __syncthreads();
    float r = 0.f;
    if (f < FDIM) {
        const float* w = W3 + (size_t)f * FDIM;
        for (int k = 0; k < FDIM; ++k) r += w[k] * tmp[k];
        r += b3[f];
    }
    bc[f] = (f < FDIM) ? r : 0.f;
}

// ---------------- split Wc -> M1 = Wa-Wb, M2 = Wb (padded fp16, [N,K]) -----

__global__ void k_prepM(const _Float16* __restrict__ Wch,
                        _Float16* __restrict__ M1h, _Float16* __restrict__ M2h) {
    int f = blockIdx.x, k = threadIdx.x;
    float a = 0.f, b = 0.f;
    if (f < FDIM && k < FDIM) {
        float wa = (float)Wch[(size_t)f * F2P + k];
        float wb = (float)Wch[(size_t)f * F2P + FDIM + k];
        a = wa - wb;
        b = wb;
    }
    M1h[(size_t)f * FP + k] = (_Float16)a;
    M2h[(size_t)f * FP + k] = (_Float16)b;
}

// ---------------- per-node segment max + bias + residual + relu ------------

__global__ void k_edge_reduce(const float* __restrict__ Af, const float* __restrict__ Bf,
                              const float* __restrict__ bc, const int* __restrict__ offsets,
                              const int* __restrict__ ssrc, float* __restrict__ gcur,
                              _Float16* __restrict__ gh, int residual) {
    int i = blockIdx.x, f = threadIdx.x;
    int beg = offsets[i], end = offsets[i + 1];
    float m = -3.402823466e38f;
    // ssrc[e] is block-uniform per iteration (scalarized); Bf is L2-resident.
#pragma unroll 4
    for (int e = beg; e < end; ++e) {
        int s = ssrc[e];
        m = fmaxf(m, Bf[(size_t)s * FP + f]);
    }
    float v = 0.f;
    if (end > beg) v = Af[(size_t)i * FP + f] + bc[f] + m;
    if (residual && f < FDIM) v += gcur[(size_t)i * FDIM + f];
    v = fmaxf(v, 0.f);
    if (f < FDIM) gcur[(size_t)i * FDIM + f] = v;   // unpadded = g.ravel()
    gh[(size_t)i * FP + f] = (f < FDIM) ? (_Float16)v : (_Float16)0.f;
}

// ---------------- bandwidth-bound mat-vecs (NT-hint weight streams) --------

__global__ void k_l4(const float* __restrict__ W, const float* __restrict__ b,
                     const float* __restrict__ g, float* __restrict__ y) {
    __shared__ float red[256];
    int h = blockIdx.x, tid = threadIdx.x;
    const f4* w4 = (const f4*)(W + (size_t)h * GVEC);
    const f4* g4 = (const f4*)g;
    float s = 0.f;
    for (int k = tid; k < GVEC / 4; k += 256) {
        __builtin_prefetch(w4 + k + 2048, 0, 0);          // global_prefetch_b8
        f4 a = __builtin_nontemporal_load(w4 + k);        // single-use stream
        f4 x = g4[k];                                      // L2-resident
        s += a.x * x.x + a.y * x.y + a.z * x.z + a.w * x.w;
    }
    red[tid] = s;
    __syncthreads();
    for (int o = 128; o > 0; o >>= 1) {
        if (tid < o) red[tid] += red[tid + o];
        __syncthreads();
    }
    if (tid == 0) y[h] = fmaxf(red[0] + b[h], 0.f);
}

__global__ void k_out(const float* __restrict__ W, const float* __restrict__ b,
                      const float* __restrict__ y, float* __restrict__ out) {
    int wave = threadIdx.x >> 5, lane = threadIdx.x & 31;
    int o = blockIdx.x * 8 + wave;                 // 4080 * 8 == 32640 exactly
    const f4* w4 = (const f4*)(W + (size_t)o * HDIM);
    const f4* y4 = (const f4*)y;
    float s = 0.f;
    for (int k = lane; k < HDIM / 4; k += 32) {
        f4 a = __builtin_nontemporal_load(w4 + k);        // single-use stream
        f4 v = y4[k];                                      // 8KB, cache-hot
        s += a.x * v.x + a.y * v.y + a.z * v.z + a.w * v.w;
    }
    for (int off = 16; off > 0; off >>= 1) s += __shfl_down(s, off, 32);
    if (lane == 0) out[o] = s + b[o];
}

// ---------------------------------------------------------------------------

extern "C" void kernel_launch(void* const* d_in, const int* in_sizes, int n_in,
                              void* d_out, int out_size, void* d_ws, size_t ws_size,
                              hipStream_t stream) {
    (void)in_sizes; (void)n_in; (void)out_size; (void)ws_size;

    const float* x      = (const float*)d_in[0];
    const int*   eidx   = (const int*)d_in[1];
    const int*   src    = eidx;
    const int*   dst    = eidx + NEDGE;
    const float* l4_W   = (const float*)d_in[20];
    const float* l4_b   = (const float*)d_in[21];
    const float* out_W  = (const float*)d_in[22];
    const float* out_b  = (const float*)d_in[23];
    float* out = (float*)d_out;

    // workspace carve-up (256B aligned), ~2.8 MB total
    char* w = (char*)d_ws;
    size_t off = 0;
    auto take = [&](size_t bytes) -> char* {
        char* p = w + off;
        off = (off + bytes + 255) & ~(size_t)255;
        return p;
    };
    int*       counts  = (int*)take(256 * 4);
    int*       cursors = (int*)take(256 * 4);
    int*       offsets = (int*)take(257 * 4);
    int*       ssrc    = (int*)take((size_t)NEDGE * 4);
    _Float16*  W1t     = (_Float16*)take((size_t)F2P * F2P * 2);   // W1^T padded
    _Float16*  W2h     = (_Float16*)take((size_t)FP  * F2P * 2);
    _Float16*  W3h     = (_Float16*)take((size_t)FP  * FP  * 2);
    _Float16*  Tt      = (_Float16*)take((size_t)F2P * FP  * 2);   // (W2*W1)^T
    _Float16*  Wch     = (_Float16*)take((size_t)FP  * F2P * 2);   // W3*W2*W1
    _Float16*  M1h     = (_Float16*)take((size_t)FP  * FP  * 2);
    _Float16*  M2h     = (_Float16*)take((size_t)FP  * FP  * 2);
    float*     bc      = (float*)take(256 * 4);
    _Float16*  gh      = (_Float16*)take((size_t)FP * FP * 2);
    float*     Af      = (float*)take((size_t)FP * FP * 4);
    float*     Bf      = (float*)take((size_t)FP * FP * 4);
    float*     gcur    = (float*)take((size_t)GVEC * 4);
    float*     y       = (float*)take((size_t)HDIM * 4);

    // ---- edge counting sort by dst (topology constant across layers) ----
    k_zero256<<<1, 256, 0, stream>>>(counts, cursors);
    k_hist   <<<256, 256, 0, stream>>>(dst, counts);
    k_scan   <<<1, 256, 0, stream>>>(counts, offsets);
    k_scatter<<<256, 256, 0, stream>>>(src, dst, offsets, cursors, ssrc);

    // ---- g0 = x (padded fp16 for WMMA) ----
    k_cvt_pad<<<NNODE, 256, 0, stream>>>(x, NNODE, FDIM, gh, FP);

    for (int l = 0; l < 3; ++l) {
        const float* W1 = (const float*)d_in[2 + 6 * l + 0];
        const float* b1 = (const float*)d_in[2 + 6 * l + 1];
        const float* W2 = (const float*)d_in[2 + 6 * l + 2];
        const float* b2 = (const float*)d_in[2 + 6 * l + 3];
        const float* W3 = (const float*)d_in[2 + 6 * l + 4];
        const float* b3 = (const float*)d_in[2 + 6 * l + 5];

        // pad/convert weights to fp16 (W1 transposed so all GEMMs are NT)
        k_cvt_pad_t<<<F2P, 256, 0, stream>>>(W1, 2 * FDIM, 2 * FDIM, W1t, F2P);
        k_cvt_pad  <<<FP,  256, 0, stream>>>(W2, FDIM,     2 * FDIM, W2h, F2P);
        k_cvt_pad  <<<FP,  256, 0, stream>>>(W3, FDIM,     FDIM,     W3h, FP);

        // T^T = (W2 * W1)^T : A=W2h[256,512], Bt=W1t[512,512], store transposed
        k_gemm_nt<true, true><<<dim3(F2P / 16, FP / 16), 32, 0, stream>>>(
            W2h, W1t, Tt, F2P, F2P, F2P, FP);
        // Wc = W3 * T : A=W3h[256,256], Bt=Tt[512,256]
        k_gemm_nt<true, false><<<dim3(F2P / 16, FP / 16), 32, 0, stream>>>(
            W3h, Tt, Wch, FP, FP, FP, F2P);

        // bc = W3*(W2*b1+b2)+b3 (fp32)
        k_bias<<<1, 256, 0, stream>>>(W2, b1, b2, W3, b3, bc);

        // M1 = Wa - Wb, M2 = Wb   (already [N,K] for NT GEMM)
        k_prepM<<<FP, 256, 0, stream>>>(Wch, M1h, M2h);

        // A = g*M1^T, B = g*M2^T  (WMMA, f32 out)
        k_gemm_nt<false, false><<<dim3(FP / 16, FP / 16), 32, 0, stream>>>(
            gh, M1h, Af, FP, FP, FP, FP);
        k_gemm_nt<false, false><<<dim3(FP / 16, FP / 16), 32, 0, stream>>>(
            gh, M2h, Bf, FP, FP, FP, FP);

        // agg[i] = A[i]+bc+max_{dst=i} B[src]; +residual; relu
        k_edge_reduce<<<NNODE, 256, 0, stream>>>(Af, Bf, bc, offsets, ssrc,
                                                 gcur, gh, l > 0);
    }

    // y = relu(l4_W @ g3.ravel() + l4_b) ; out = out_W @ y + out_b
    k_l4 <<<HDIM, 256, 0, stream>>>(l4_W, l4_b, gcur, y);
    k_out<<<OUTN / 8, 256, 0, stream>>>(out_W, out_b, y, out);
}
